// HyperINR_2740189135424
// MI455X (gfx1250) — compile-verified, matrix operand order checked by reading, not run on hardware
//
#include <hip/hip_runtime.h>
#include <hip/hip_bf16.h>

typedef _Float16 f16;
typedef _Float16 v8h  __attribute__((ext_vector_type(8)));
typedef _Float16 v16h __attribute__((ext_vector_type(16)));
typedef float    v8f  __attribute__((ext_vector_type(8)));

#define OMEGA 30.0f
#define HID   128
#define PTS   128
#define NPTS  32768
#define NB    16
#define PPAR  50177

// offsets (floats) into one sample's param vector, state_dict order
#define OFF_W0 0
#define OFF_B0 384
#define OFF_W1 512
#define OFF_B1 16896
#define OFF_W2 17024
#define OFF_B2 33408
#define OFF_W3 33536
#define OFF_B3 49920
#define OFF_WO 50048
#define OFF_BO 50176

// ---------------------------------------------------------------------------
// Kernel A: hypernet hidden layers (tiny, one workgroup). h2 [16][256] -> ws
// ---------------------------------------------------------------------------
__global__ __launch_bounds__(256) void hyper_hidden(
    const float* __restrict__ t,
    const float* __restrict__ hW0, const float* __restrict__ hb0,
    const float* __restrict__ hW1, const float* __restrict__ hb1,
    const float* __restrict__ hW2, const float* __restrict__ hb2,
    float* __restrict__ h2out) {
  __shared__ float tl[NB][8];
  __shared__ float h0[NB][256];
  __shared__ float h1[NB][256];
  const int tid = threadIdx.x;
  if (tid < NB * 8) tl[tid >> 3][tid & 7] = t[tid];
  __syncthreads();

  const int o = tid;  // output neuron, 0..255
  float acc[NB];

  // layer 0: 8 -> 256
  {
    float bv = hb0[o];
    #pragma unroll
    for (int b = 0; b < NB; ++b) acc[b] = bv;
    #pragma unroll
    for (int i = 0; i < 8; ++i) {
      float w = hW0[o * 8 + i];
      #pragma unroll
      for (int b = 0; b < NB; ++b) acc[b] += w * tl[b][i];
    }
    #pragma unroll
    for (int b = 0; b < NB; ++b) h0[b][o] = __sinf(OMEGA * acc[b]);
  }
  __syncthreads();

  // layer 1: 256 -> 256
  {
    float bv = hb1[o];
    #pragma unroll
    for (int b = 0; b < NB; ++b) acc[b] = bv;
    for (int k = 0; k < 256; ++k) {
      float w = hW1[o * 256 + k];
      #pragma unroll
      for (int b = 0; b < NB; ++b) acc[b] += w * h0[b][k];
    }
    #pragma unroll
    for (int b = 0; b < NB; ++b) h1[b][o] = __sinf(OMEGA * acc[b]);
  }
  __syncthreads();

  // layer 2: 256 -> 256, write to workspace
  {
    float bv = hb2[o];
    #pragma unroll
    for (int b = 0; b < NB; ++b) acc[b] = bv;
    for (int k = 0; k < 256; ++k) {
      float w = hW2[o * 256 + k];
      #pragma unroll
      for (int b = 0; b < NB; ++b) acc[b] += w * h1[b][k];
    }
    #pragma unroll
    for (int b = 0; b < NB; ++b) h2out[b * 256 + o] = __sinf(OMEGA * acc[b]);
  }
}

// ---------------------------------------------------------------------------
// Kernel B: hypernet output layer. params[b][j] = h2[b] . hWo[j] + hbo[j]
// Bandwidth bound on streaming hWo (51 MB). 16 accumulators per thread.
// ---------------------------------------------------------------------------
__global__ __launch_bounds__(256) void hyper_out(
    const float* __restrict__ h2, const float* __restrict__ hWo,
    const float* __restrict__ hbo, float* __restrict__ params) {
  __shared__ float hl[NB][256];
  const int tid = threadIdx.x;
  for (int i = tid; i < NB * 256; i += 256) hl[i >> 8][i & 255] = h2[i];
  __syncthreads();

  const int j = blockIdx.x * 256 + tid;
  if (j >= PPAR) return;
  float acc[NB];
  float bv = hbo[j];
  #pragma unroll
  for (int b = 0; b < NB; ++b) acc[b] = bv;
  const float* wr = hWo + (size_t)j * 256;
  for (int k = 0; k < 256; ++k) {
    float w = wr[k];
    #pragma unroll
    for (int b = 0; b < NB; ++b) acc[b] += w * hl[b][k];
  }
  #pragma unroll
  for (int b = 0; b < NB; ++b) params[(size_t)b * PPAR + j] = acc[b];
}

// ---------------------------------------------------------------------------
// Kernel C: per-sample SIREN over 128-point tiles. Hidden layers via
// v_wmma_f32_16x16x32_f16. Block = 256 threads = 8 waves; wave w owns
// output columns [16w, 16w+16). Activations and weights staged in LDS (f16).
// ---------------------------------------------------------------------------
__global__ __launch_bounds__(256) void inr_eval(
    const float* __restrict__ xt, const float* __restrict__ params,
    float* __restrict__ out) {
  __shared__ __align__(16) f16 hA[PTS][HID];
  __shared__ __align__(16) f16 hB[PTS][HID];
  __shared__ __align__(16) f16 Wl[HID][HID];   // W[out][in], f16
  __shared__ float biasL[HID];
  __shared__ float xl[PTS][4];
  __shared__ float w0l[HID][3];
  __shared__ float wol[HID];

  const int tid   = threadIdx.x;
  const int b     = blockIdx.x >> 8;
  const int pbase = (blockIdx.x & 255) * PTS;
  const float* pp = params + (size_t)b * PPAR;

  // ---- stage query points + layer-0 weights ----
  for (int i = tid; i < PTS * 3; i += 256)
    xl[i / 3][i % 3] = xt[((size_t)b * NPTS + pbase) * 3 + i];
  for (int i = tid; i < HID * 3; i += 256)
    w0l[i / 3][i % 3] = pp[OFF_W0 + i];
  if (tid < HID) biasL[tid] = pp[OFF_B0 + tid];
  __syncthreads();

  // ---- layer 0 (K=3): VALU ----
  for (int idx = tid; idx < PTS * HID; idx += 256) {
    int n = idx >> 7, o = idx & 127;
    float s = biasL[o] + xl[n][0] * w0l[o][0] + xl[n][1] * w0l[o][1] +
              xl[n][2] * w0l[o][2];
    hA[n][o] = (f16)__sinf(OMEGA * s);
  }
  __syncthreads();

  const int lane  = tid & 31;
  const int wv    = tid >> 5;          // wave id = N-tile
  const int mrow  = lane & 15;         // row within tile (A) / col (B,D)
  const int khalf = (lane >> 4) * 8;   // 0 for lanes 0-15, 8 for lanes 16-31
  const int drow  = (lane >> 4) * 8;   // D-matrix row offset for VGPR g

  const int Woff[3] = {OFF_W1, OFF_W2, OFF_W3};
  const int Boff[3] = {OFF_B1, OFF_B2, OFF_B3};
  f16 (*src)[HID] = hA;
  f16 (*dst)[HID] = hB;

  union HV { v16h v; v8h h[2]; };

  for (int L = 0; L < 3; ++L) {
    // convert this layer's weights f32->f16 into LDS, natural [out][in] order
    for (int idx = tid; idx < HID * HID; idx += 256)
      Wl[idx >> 7][idx & 127] = (f16)pp[Woff[L] + idx];
    if (tid < HID) biasL[tid] = pp[Boff[L] + tid];
    __syncthreads();

    // preload this wave's 4 B fragments (K-chunks of 32) — 2x ds_load_b128 each
    HV bf[4];
    const f16* wrow = &Wl[wv * 16 + mrow][0];
    #pragma unroll
    for (int kk = 0; kk < 4; ++kk) {
      bf[kk].h[0] = *(const v8h*)(wrow + kk * 32 + khalf);
      bf[kk].h[1] = *(const v8h*)(wrow + kk * 32 + 16 + khalf);
    }
    const float biasv = biasL[wv * 16 + mrow];

    #pragma unroll
    for (int mt = 0; mt < 8; ++mt) {
      v8f acc;
      #pragma unroll
      for (int g = 0; g < 8; ++g) acc[g] = biasv;  // bias constant per column
      const f16* arow = &src[mt * 16 + mrow][0];
      #pragma unroll
      for (int kk = 0; kk < 4; ++kk) {
        HV af;
        af.h[0] = *(const v8h*)(arow + kk * 32 + khalf);
        af.h[1] = *(const v8h*)(arow + kk * 32 + 16 + khalf);
        acc = __builtin_amdgcn_wmma_f32_16x16x32_f16(
            false, af.v, false, bf[kk].v, (short)0, acc, false, false);
      }
      // fused activation: D element (M = mt*16+drow+g, N = wv*16+mrow)
      #pragma unroll
      for (int g = 0; g < 8; ++g)
        dst[mt * 16 + drow + g][wv * 16 + mrow] = (f16)__sinf(OMEGA * acc[g]);
    }
    __syncthreads();
    f16 (*tmp)[HID] = src; src = dst; dst = tmp;
  }
  // after 3 layers final activations are in `src` (== hB)

  // ---- output layer (N=1): VALU dot products ----
  if (tid < HID) wol[tid] = pp[OFF_WO + tid];
  __syncthreads();
  if (tid < PTS) {
    float s = pp[OFF_BO];
    const f16* hrow = &src[tid][0];
    #pragma unroll 8
    for (int i = 0; i < HID; ++i) s += (float)hrow[i] * wol[i];
    out[(size_t)b * NPTS + pbase + tid] = s;
  }
}

// ---------------------------------------------------------------------------
extern "C" void kernel_launch(void* const* d_in, const int* in_sizes, int n_in,
                              void* d_out, int out_size, void* d_ws, size_t ws_size,
                              hipStream_t stream) {
  const float* t   = (const float*)d_in[0];
  const float* xt  = (const float*)d_in[1];
  const float* hW0 = (const float*)d_in[2];
  const float* hb0 = (const float*)d_in[3];
  const float* hW1 = (const float*)d_in[4];
  const float* hb1 = (const float*)d_in[5];
  const float* hW2 = (const float*)d_in[6];
  const float* hb2 = (const float*)d_in[7];
  const float* hWo = (const float*)d_in[8];
  const float* hbo = (const float*)d_in[9];
  float* out = (float*)d_out;

  float* ws     = (float*)d_ws;
  float* h2     = ws;                 // 16*256 floats
  float* params = ws + NB * 256;      // 16*50177 floats (~3.2 MB, fits L2)

  hipLaunchKernelGGL(hyper_hidden, dim3(1), dim3(256), 0, stream,
                     t, hW0, hb0, hW1, hb1, hW2, hb2, h2);
  hipLaunchKernelGGL(hyper_out, dim3((PPAR + 255) / 256), dim3(256), 0, stream,
                     h2, hWo, hbo, params);
  hipLaunchKernelGGL(inr_eval, dim3(NB * 256), dim3(256), 0, stream,
                     xt, params, out);
}
